// GCNActorCritic_51376398795484
// MI455X (gfx1250) — matrix operand order repaired and checked by reading, not compiled, not run on hardware
//
#include <hip/hip_runtime.h>

// ---------------------------------------------------------------------------
// GATv2 actor-critic for MI455X (gfx1250), wave32 + WMMA bf16 + TDM.
//
// Assumed d_in ordering: top-level dict insertion order (x, edge_index, batch,
// current_device_idx), then params flattened as a JAX pytree (dict keys
// sorted, recursively):
//   4:aW1 5:aW2 6:ab1 7:ab2
//   8..31:  actor_bb  = c1{Wl,Wr,att,bias,bl,br} c2{...} c3{...}
//                       ln1b ln1g ln2b ln2g ln3b ln3g
//   32:cW1 33:cW2 34:cb1 35:cb2
//   36..59: critic_bb (same layout as actor_bb)
// ---------------------------------------------------------------------------

#define NN       20000
#define NE       400000
#define NG       128
#define FIN      16
#define HIDN     128
#define NACT     256
#define ETOT     (NE + NN)
#define ENC_NEG_INF 0x007FFFFFu   // encodeF(-inf)

typedef __attribute__((ext_vector_type(16))) __bf16 v16bf;
typedef __attribute__((ext_vector_type(8)))  __bf16 v8bf;
typedef __attribute__((ext_vector_type(8)))  float  v8f;
typedef __attribute__((ext_vector_type(4)))  unsigned v4u;
typedef __attribute__((ext_vector_type(8)))  int    v8i;
typedef __attribute__((ext_vector_type(4)))  int    v4i;

// ---- order-preserving float<->uint encoding for atomic max over floats ----
__device__ __forceinline__ unsigned encF(float f) {
  unsigned u = __float_as_uint(f);
  return (u & 0x80000000u) ? ~u : (u | 0x80000000u);
}
__device__ __forceinline__ float decF(unsigned k) {
  return (k & 0x80000000u) ? __uint_as_float(k ^ 0x80000000u)
                           : __uint_as_float(~k);
}

// ---------------------------------------------------------------------------
// WMMA GEMM:  C[M,NC] = A[M,KP](bf16 row-major) * Bt[NC,KP](bf16, transposed)
//             + bias[NC], optional ReLU.
// Block = 8 waves; block stages the whole Bt panel (NC*KP bf16 <= 64KB) into
// LDS via the Tensor Data Mover, then each wave owns one 16-row strip of A
// (fragments hoisted into registers) and sweeps all NC/16 tiles from LDS.
// NC compile-time => C stores become immediate-offset stores off one base.
// ---------------------------------------------------------------------------
template<int KP, int NC>
__global__ __launch_bounds__(256)
void gemm_bf16_wmma(const __bf16* __restrict__ A, const __bf16* __restrict__ Bt,
                    const float* __restrict__ bias, float* __restrict__ C,
                    int M, int relu) {
  static_assert(KP * NC <= 32768, "B panel must fit 64KB LDS");
  __shared__ __attribute__((aligned(64))) __bf16 Bs[KP * NC];
  const int lane   = threadIdx.x & 31;
  const int waveId = threadIdx.x >> 5;
  const int lh = lane & 15;
  const int hi = lane >> 4;
  constexpr unsigned total = (unsigned)(NC * KP);   // elements

  if (waveId == 0) {
    // ---- TDM: 1-D tile of `total` bf16 elements  global -> LDS ----
    unsigned ldsA = (unsigned)(size_t)(&Bs[0]);     // low 32 bits = LDS offset
    unsigned alo  = (unsigned)(size_t)Bt;
    unsigned ahi  = (unsigned)((size_t)Bt >> 32);
    v4u g0; v8i g1;
    g0[0] = 1u;                                     // count=1, user mode
    g0[1] = ldsA;                                   // lds_addr
    g0[2] = alo;                                    // global_addr[31:0]
    g0[3] = (ahi & 0x01FFFFFFu) | (2u << 30);       // global_addr[56:32] | type=2
    g1[0] = (int)(1u << 16);                        // data_size = 2 bytes
    g1[1] = (int)((total & 0xFFFFu) << 16);         // tensor_dim0[15:0]
    g1[2] = (int)(((total >> 16) & 0xFFFFu) | (1u << 16)); // dim0[31:16]|dim1=1
    g1[3] = (int)((total & 0xFFFFu) << 16);         // tile_dim0 = total
    g1[4] = 0;                                      // tile_dim1=0, tile_dim2=0
    g1[5] = (int)total;                             // tensor_dim0_stride[31:0]
    g1[6] = 0;
    g1[7] = 0;
#if __clang_major__ >= 23
    __builtin_amdgcn_tensor_load_to_lds(g0, g1, (v4i)0, (v4i)0, (v8i)0, 0);
#else
    __builtin_amdgcn_tensor_load_to_lds(g0, g1, (v4i)0, (v4i)0, 0);
#endif
    __builtin_amdgcn_s_wait_tensorcnt(0);
  }
  __syncthreads();

  const int mT    = M >> 4;
  const int mTile = blockIdx.x * 8 + waveId;
  const bool active = (mTile < mT);                 // wave-uniform
  if (active) {
    const int m0 = mTile << 4;
    // A (16x32 bf16 frag): lanes 0-15 K=[k0..k0+15];
    //                      lanes 16-31 K=[k0+16..k0+31]
    const __bf16* arow = A + (size_t)(m0 + lh) * KP;
    v8bf a0[KP / 32], a1[KP / 32];
#pragma unroll
    for (int kk = 0; kk < KP / 32; ++kk) {
      a0[kk] = *(const v8bf*)(arow + kk * 32 + hi * 16);
      a1[kk] = *(const v8bf*)(arow + kk * 32 + hi * 16 + 8);
    }
#pragma unroll
    for (int nt = 0; nt < NC / 16; ++nt) {
      v8f acc = {};
#pragma unroll
      for (int kk = 0; kk < KP / 32; ++kk) {
        union { v16bf v; v8bf h[2]; } ua, ub;
        ua.h[0] = a0[kk]; ua.h[1] = a1[kk];
        // B (32x16): lane holds column nt*16+lh, 16 contiguous K values
        ub.v = *(const v16bf*)(&Bs[(nt * 16 + lh) * KP + kk * 32 + hi * 16]);
        acc = __builtin_amdgcn_wmma_f32_16x16x32_bf16(false, ua.v, false, ub.v,
                                                      (short)0, acc, false, false);
      }
      const int col = (nt << 4) + lh;
      const float bv = bias[col];
      float* crow = C + (size_t)(m0 + hi * 8) * NC + col;
#pragma unroll
      for (int i = 0; i < 8; ++i) {
        float v = acc[i] + bv;
        if (relu) v = fmaxf(v, 0.0f);
        crow[i * NC] = v;                           // immediate-offset stores
      }
    }
  }
}

// f32 [rows,K] -> bf16 [rows,Kp], zero-padded K
__global__ void cast_pad_k(const float* __restrict__ src, __bf16* __restrict__ dst,
                           int rows, int K, int Kp) {
  int t = blockIdx.x * blockDim.x + threadIdx.x;
  if (t >= rows * Kp) return;
  int r = t / Kp, k = t - r * Kp;
  dst[t] = (__bf16)(k < K ? src[(size_t)r * K + k] : 0.0f);
}

// W f32 [K,N] -> Bt bf16 [N,Kp] (transpose + cast + pad)
__global__ void wt_cast_k(const float* __restrict__ W, __bf16* __restrict__ Bt,
                          int K, int N, int Kp) {
  int t = blockIdx.x * blockDim.x + threadIdx.x;
  if (t >= N * Kp) return;
  int n = t / Kp, k = t - n * Kp;
  Bt[t] = (__bf16)(k < K ? W[(size_t)k * N + n] : 0.0f);
}

__global__ void init_m_k(unsigned* __restrict__ m, int n) {
  int t = blockIdx.x * blockDim.x + threadIdx.x;
  if (t < n) m[t] = ENC_NEG_INF;
}

// edge id e: e < NE -> (src,dst) from edge_index; else self-loop (e-NE, e-NE)
__device__ __forceinline__ void edge_sd(const int* ei, int e, int& s, int& d) {
  if (e < NE) { s = ei[e]; d = ei[NE + e]; } else { s = d = e - NE; }
}

// lg[e,h] = att[h,:] . leaky_relu(xl[s,h,:] + xr[d,h,:]); segment max into mEnc
__global__ void edge_logits_k(const float* __restrict__ xl, const float* __restrict__ xr,
                              const int* __restrict__ ei, const float* __restrict__ att,
                              float* __restrict__ lg, unsigned* __restrict__ mEnc,
                              int H, int C) {
  int t = blockIdx.x * blockDim.x + threadIdx.x;
  if (t >= ETOT * H) return;
  int e = t / H, h = t - e * H;
  int s, d; edge_sd(ei, e, s, d);
  const float* a = xl + (size_t)s * HIDN + h * C;
  const float* b = xr + (size_t)d * HIDN + h * C;
  const float* w = att + h * C;
  float acc = 0.0f;
  for (int c = 0; c < C; ++c) {
    float v = a[c] + b[c];
    v = v > 0.0f ? v : 0.2f * v;
    acc += w[c] * v;
  }
  lg[(size_t)e * H + h] = acc;
  atomicMax(&mEnc[(size_t)d * H + h], encF(acc));
}

// a = exp(lg - m[dst]); lg <- a; z[dst] += a
__global__ void edge_exp_k(float* __restrict__ lg, const unsigned* __restrict__ mEnc,
                           float* __restrict__ z, const int* __restrict__ ei, int H) {
  int t = blockIdx.x * blockDim.x + threadIdx.x;
  if (t >= ETOT * H) return;
  int e = t / H, h = t - e * H;
  int s, d; edge_sd(ei, e, s, d); (void)s;
  float a = __expf(lg[t] - decF(mEnc[(size_t)d * H + h]));
  lg[t] = a;
  atomicAdd(&z[(size_t)d * H + h], a);
}

// acc[dst,h,:] += (a/z[dst,h]) * xl[src,h,:]
__global__ void edge_accum_k(const float* __restrict__ lg, const float* __restrict__ z,
                             const float* __restrict__ xl, const int* __restrict__ ei,
                             float* __restrict__ accb, int H, int C) {
  int t = blockIdx.x * blockDim.x + threadIdx.x;
  if (t >= ETOT * H) return;
  int e = t / H, h = t - e * H;
  int s, d; edge_sd(ei, e, s, d);
  float w = lg[t] / z[(size_t)d * H + h];
  const float* src = xl + (size_t)s * HIDN + h * C;
  float* dst = accb + (size_t)d * HIDN + h * C;
  for (int c = 0; c < C; ++c) atomicAdd(&dst[c], w * src[c]);
}

// y = relu(LN(acc + conv_bias) * g + b); write f32 and bf16 copies
__global__ __launch_bounds__(128)
void bias_ln_relu_k(const float* __restrict__ accb, const float* __restrict__ cbias,
                    const float* __restrict__ g, const float* __restrict__ b,
                    float* __restrict__ hOut, __bf16* __restrict__ hBf) {
  int nd = blockIdx.x, f = threadIdx.x;
  float v = accb[(size_t)nd * HIDN + f] + cbias[f];
  __shared__ float red[HIDN];
  red[f] = v; __syncthreads();
  for (int s = 64; s > 0; s >>= 1) { if (f < s) red[f] += red[f + s]; __syncthreads(); }
  float mu = red[0] * (1.0f / HIDN); __syncthreads();
  float dlt = v - mu;
  red[f] = dlt * dlt; __syncthreads();
  for (int s = 64; s > 0; s >>= 1) { if (f < s) red[f] += red[f + s]; __syncthreads(); }
  float var = red[0] * (1.0f / HIDN);
  float y = dlt * rsqrtf(var + 1e-5f) * g[f] + b[f];
  y = fmaxf(y, 0.0f);
  hOut[(size_t)nd * HIDN + f] = y;
  hBf [(size_t)nd * HIDN + f] = (__bf16)y;
}

__global__ void count_nodes_k(const int* __restrict__ batch, float* __restrict__ cnt) {
  int t = blockIdx.x * blockDim.x + threadIdx.x;
  if (t < NN) atomicAdd(&cnt[batch[t]], 1.0f);
}
__global__ void pool_accum_k(const float* __restrict__ h, const int* __restrict__ batch,
                             float* __restrict__ pool) {
  int t = blockIdx.x * blockDim.x + threadIdx.x;
  if (t >= NN * HIDN) return;
  int nd = t / HIDN, f = t - nd * HIDN;
  atomicAdd(&pool[(size_t)batch[nd] * HIDN + f], h[t]);
}
__global__ void pool_div_k(float* __restrict__ pool, const float* __restrict__ cnt) {
  int t = blockIdx.x * blockDim.x + threadIdx.x;
  if (t < NG * HIDN) pool[t] /= fmaxf(cnt[t / HIDN], 1.0f);
}
__global__ void build_actor_in_k(const float* __restrict__ pool, const float* __restrict__ h,
                                 const int* __restrict__ cdi, float* __restrict__ ain) {
  int t = blockIdx.x * blockDim.x + threadIdx.x;
  if (t >= NG * 2 * HIDN) return;
  int gph = t / (2 * HIDN), f = t - gph * 2 * HIDN;
  ain[t] = (f < HIDN) ? pool[(size_t)gph * HIDN + f]
                      : h[(size_t)cdi[gph] * HIDN + (f - HIDN)];
}
__global__ __launch_bounds__(128)
void critic_out_k(const float* __restrict__ hid, const float* __restrict__ cW2,
                  const float* __restrict__ cb2, float* __restrict__ out) {
  int gph = blockIdx.x, f = threadIdx.x;
  __shared__ float red[HIDN];
  red[f] = hid[(size_t)gph * HIDN + f] * cW2[f]; __syncthreads();
  for (int s = 64; s > 0; s >>= 1) { if (f < s) red[f] += red[f + s]; __syncthreads(); }
  if (f == 0) out[gph] = red[0] + cb2[0];
}

// ---------------------------------------------------------------------------
extern "C" void kernel_launch(void* const* d_in, const int* in_sizes, int n_in,
                              void* d_out, int out_size, void* d_ws, size_t ws_size,
                              hipStream_t stream) {
  auto F32 = [&](int i) { return (const float*)d_in[i]; };
  const float* x    = F32(0);
  const int* ei     = (const int*)d_in[1];
  const int* batch  = (const int*)d_in[2];
  const int* cdi    = (const int*)d_in[3];

  // ---- workspace suballocation (256B aligned) ----
  char* base = (char*)d_ws; size_t off = 0;
  auto alloc = [&](size_t bytes) { off = (off + 255) & ~(size_t)255;
                                   void* p = base + off; off += bytes; return p; };
  __bf16*   xbf   = (__bf16*)  alloc((size_t)NN * HIDN * 2);
  float*    xl    = (float*)   alloc((size_t)NN * HIDN * 4);
  float*    xr    = (float*)   alloc((size_t)NN * HIDN * 4);
  float*    accb  = (float*)   alloc((size_t)NN * HIDN * 4);
  float*    hbuf  = (float*)   alloc((size_t)NN * HIDN * 4);
  float*    lg    = (float*)   alloc((size_t)ETOT * 4 * 4);
  unsigned* mEnc  = (unsigned*)alloc((size_t)NN * 4 * 4);
  float*    zb    = (float*)   alloc((size_t)NN * 4 * 4);
  float*    pool  = (float*)   alloc((size_t)NG * HIDN * 4);
  float*    cnt   = (float*)   alloc((size_t)NG * 4);
  float*    ain   = (float*)   alloc((size_t)NG * 2 * HIDN * 4);
  __bf16*   ainbf = (__bf16*)  alloc((size_t)NG * 2 * HIDN * 2);
  float*    hid1  = (float*)   alloc((size_t)NG * HIDN * 4);
  __bf16*   hidbf = (__bf16*)  alloc((size_t)NG * HIDN * 2);
  __bf16*   wbfA  = (__bf16*)  alloc((size_t)HIDN * 2 * HIDN * 2);
  __bf16*   wbfB  = (__bf16*)  alloc((size_t)NACT * HIDN * 2);

  auto gemm = [&](const __bf16* A, const __bf16* Bt, const float* bias, float* Cp,
                  int M, int N, int Kp, int relu) {
    int blocks = ((M >> 4) + 7) / 8;
    if (Kp == 32 && N == 128)
      gemm_bf16_wmma<32, 128><<<blocks, 256, 0, stream>>>(A, Bt, bias, Cp, M, relu);
    else if (Kp == 128 && N == 128)
      gemm_bf16_wmma<128, 128><<<blocks, 256, 0, stream>>>(A, Bt, bias, Cp, M, relu);
    else if (Kp == 256 && N == 128)
      gemm_bf16_wmma<256, 128><<<blocks, 256, 0, stream>>>(A, Bt, bias, Cp, M, relu);
    else
      gemm_bf16_wmma<128, 256><<<blocks, 256, 0, stream>>>(A, Bt, bias, Cp, M, relu);
  };
  auto grid = [](int n) { return (n + 255) / 256; };

  // one GATv2 conv layer + LN + ReLU; input activation lives in xbf (bf16)
  auto run_layer = [&](int pbase, const float* lng, const float* lnb,
                       int H, int C, int din, bool first) {
    const float *Wl = F32(pbase), *Wr = F32(pbase + 1), *att = F32(pbase + 2);
    const float *cbias = F32(pbase + 3), *bl = F32(pbase + 4), *br = F32(pbase + 5);
    int Kp = (din == FIN) ? 32 : HIDN;
    wt_cast_k<<<grid(HIDN * Kp), 256, 0, stream>>>(Wl, wbfA, din, HIDN, Kp);
    wt_cast_k<<<grid(HIDN * Kp), 256, 0, stream>>>(Wr, wbfB, din, HIDN, Kp);
    if (first)
      cast_pad_k<<<grid(NN * 32), 256, 0, stream>>>(x, xbf, NN, FIN, 32);
    gemm(xbf, wbfA, bl, xl, NN, HIDN, Kp, 0);
    gemm(xbf, wbfB, br, xr, NN, HIDN, Kp, 0);
    hipMemsetAsync(accb, 0, (size_t)NN * HIDN * 4, stream);
    hipMemsetAsync(zb,   0, (size_t)NN * H * 4,    stream);
    init_m_k<<<grid(NN * H), 256, 0, stream>>>(mEnc, NN * H);
    int EH = ETOT * H;
    edge_logits_k<<<grid(EH), 256, 0, stream>>>(xl, xr, ei, att, lg, mEnc, H, C);
    edge_exp_k  <<<grid(EH), 256, 0, stream>>>(lg, mEnc, zb, ei, H);
    edge_accum_k<<<grid(EH), 256, 0, stream>>>(lg, zb, xl, ei, accb, H, C);
    bias_ln_relu_k<<<NN, 128, 0, stream>>>(accb, cbias, lng, lnb, hbuf, xbf);
  };

  auto run_backbone = [&](int bb) {   // bb = index of c1.Wl
    run_layer(bb + 0,  F32(bb + 19), F32(bb + 18), 4, 32, FIN,  true);
    run_layer(bb + 6,  F32(bb + 21), F32(bb + 20), 4, 32, HIDN, false);
    run_layer(bb + 12, F32(bb + 23), F32(bb + 22), 1, HIDN, HIDN, false);
  };
  auto run_pool = [&]() {
    hipMemsetAsync(pool, 0, (size_t)NG * HIDN * 4, stream);
    hipMemsetAsync(cnt,  0, (size_t)NG * 4, stream);
    count_nodes_k<<<grid(NN), 256, 0, stream>>>(batch, cnt);
    pool_accum_k<<<grid(NN * HIDN), 256, 0, stream>>>(hbuf, batch, pool);
    pool_div_k<<<grid(NG * HIDN), 256, 0, stream>>>(pool, cnt);
  };

  // ================= actor =================
  run_backbone(8);            // actor_bb leaves start at d_in[8]
  run_pool();
  build_actor_in_k<<<grid(NG * 2 * HIDN), 256, 0, stream>>>(pool, hbuf, cdi, ain);
  cast_pad_k<<<grid(NG * 2 * HIDN), 256, 0, stream>>>(ain, ainbf, NG, 2 * HIDN, 2 * HIDN);
  wt_cast_k<<<grid(HIDN * 2 * HIDN), 256, 0, stream>>>(F32(4), wbfA, 2 * HIDN, HIDN, 2 * HIDN);
  gemm(ainbf, wbfA, F32(6), hid1, NG, HIDN, 2 * HIDN, 1);                // relu(ain@aW1+ab1)
  cast_pad_k<<<grid(NG * HIDN), 256, 0, stream>>>(hid1, hidbf, NG, HIDN, HIDN);
  wt_cast_k<<<grid(NACT * HIDN), 256, 0, stream>>>(F32(5), wbfB, HIDN, NACT, HIDN);
  gemm(hidbf, wbfB, F32(7), (float*)d_out, NG, NACT, HIDN, 0);           // logits -> d_out

  // ================= critic =================
  run_backbone(36);           // critic_bb leaves start at d_in[36]
  run_pool();
  cast_pad_k<<<grid(NG * HIDN), 256, 0, stream>>>(pool, ainbf, NG, HIDN, HIDN);
  wt_cast_k<<<grid(HIDN * HIDN), 256, 0, stream>>>(F32(32), wbfA, HIDN, HIDN, HIDN);
  gemm(ainbf, wbfA, F32(34), hid1, NG, HIDN, HIDN, 1);                   // relu(pool@cW1+cb1)
  critic_out_k<<<NG, 128, 0, stream>>>(hid1, F32(33), F32(35),
                                       (float*)d_out + NG * NACT);       // values
  (void)in_sizes; (void)n_in; (void)out_size; (void)ws_size;
}
